// ContextualAttention_85366769975388
// MI455X (gfx1250) — compile-verified, matrix operand order checked by reading, not run on hardware
//
#include <hip/hip_runtime.h>

#define HW 64
#define L  4096      // HW*HW
#define C  64
#define KP 576       // C*9

typedef __attribute__((ext_vector_type(16))) _Float16 v16h;
typedef __attribute__((ext_vector_type(8)))  _Float16 v8h;
typedef __attribute__((ext_vector_type(8)))  float    v8f;

// (y,x) <-> (x,y) flat-index permutation for 64x64 (involution)
__device__ __forceinline__ int permYX(int i) { return ((i & 63) << 6) | (i >> 6); }

// -------------------------------------------------------------------------
// Prep: build fpatch (Lx576 f16, zero pad), bsi (Lx576 f16, normalized bg
// patches, replicate pad), biT (576xL f16, un-normalized bg patches
// transposed), mi (L f32 validity mask). One block per spatial position l,
// one thread per channel.
// -------------------------------------------------------------------------
__global__ void __launch_bounds__(64)
prep_kernel(const float* __restrict__ f, const float* __restrict__ bsrc,
            const float* __restrict__ mask,
            _Float16* __restrict__ fpatch, _Float16* __restrict__ bsi,
            _Float16* __restrict__ biT, float* __restrict__ mi)
{
    __shared__ float red[64];
    const int l  = blockIdx.x;
    const int y  = l >> 6, x = l & 63;
    const int ch = threadIdx.x;

    float bv[9];
    float ssq = 0.f;
#pragma unroll
    for (int p = 0; p < 9; ++p) {
        const int dy = p / 3 - 1, dx = p % 3 - 1;
        const int yy = y + dy, xx = x + dx;
        float fv = 0.f;
        if (yy >= 0 && yy < HW && xx >= 0 && xx < HW)
            fv = f[(size_t)ch * L + yy * HW + xx];
        const int yc = min(max(yy, 0), HW - 1);
        const int xc = min(max(xx, 0), HW - 1);
        const float bb = bsrc[(size_t)ch * L + yc * HW + xc];
        bv[p] = bb;
        ssq += bb * bb;
        fpatch[(size_t)l * KP + ch * 9 + p]  = (_Float16)fv;
        biT[(size_t)(ch * 9 + p) * L + l]    = (_Float16)bb;
    }
    red[ch] = ssq;
    __syncthreads();
    for (int s = 32; s > 0; s >>= 1) {
        if (ch < s) red[ch] += red[ch + s];
        __syncthreads();
    }
    const float inv = 1.0f / fmaxf(sqrtf(red[0]), 1e-4f);
#pragma unroll
    for (int p = 0; p < 9; ++p)
        bsi[(size_t)l * KP + ch * 9 + p] = (_Float16)(bv[p] * inv);

    if (ch == 0) {
        float s = 0.f;
        for (int dy = -1; dy <= 1; ++dy)
            for (int dx = -1; dx <= 1; ++dx) {
                const int yy = y + dy, xx = x + dx;
                if (yy >= 0 && yy < HW && xx >= 0 && xx < HW)
                    s += mask[yy * HW + xx];
            }
        mi[l] = (s == 0.f) ? 1.f : 0.f;
    }
}

// -------------------------------------------------------------------------
// Fragment loader: 16 rows x 32 K of a row-major (rows x ld) f16 matrix,
// laid out per the CDNA5 16-bit A-fragment table:
//   lanes 0-15  : row = row0+lane,    K = k0+{0..7}  and k0+{16..23}
//   lanes 16-31 : row = row0+lane-16, K = k0+{8..15} and k0+{24..31}
// Two 16B global_load_b128 per lane.
// -------------------------------------------------------------------------
__device__ __forceinline__ v16h load_frag(const _Float16* __restrict__ X,
                                          int row0, int ld, int k0, int lane)
{
    const int r  = row0 + (lane & 15);
    const int kb = k0 + ((lane >> 4) << 3);
    const _Float16* p = X + (size_t)r * ld + kb;
    const v8h lo = *(const v8h*)(p);
    const v8h hi = *(const v8h*)(p + 16);
    v16h out;
#pragma unroll
    for (int i = 0; i < 8; ++i) { out[i] = lo[i]; out[i + 8] = hi[i]; }
    return out;
}

// -------------------------------------------------------------------------
// C (MxN f32, ldC) = A (MxK f16, ldA) * B^T, with B stored row-major NxK.
// Block: 256 threads = 8 waves, tile 128 x (WN*32); wave = 32 x (WN*16).
// -------------------------------------------------------------------------
template <int WN>
__global__ void __launch_bounds__(256)
gemm_nt(const _Float16* __restrict__ A, const _Float16* __restrict__ B,
        float* __restrict__ Cmat, int M, int N, int K,
        int ldA, int ldB, int ldC)
{
    const int lane = threadIdx.x & 31;
    const int wave = threadIdx.x >> 5;
    const int wm   = wave >> 1;          // 0..3
    const int wn   = wave & 1;           // 0..1
    const int m0   = blockIdx.x * 128 + wm * 32;
    const int n0   = blockIdx.y * (WN * 32) + wn * (WN * 16);

    v8f acc[2][WN];
#pragma unroll
    for (int i = 0; i < 2; ++i)
#pragma unroll
        for (int j = 0; j < WN; ++j)
#pragma unroll
            for (int e = 0; e < 8; ++e) acc[i][j][e] = 0.f;

    for (int k0 = 0; k0 < K; k0 += 32) {
        const v16h a0 = load_frag(A, m0,      ldA, k0, lane);
        const v16h a1 = load_frag(A, m0 + 16, ldA, k0, lane);
#pragma unroll
        for (int j = 0; j < WN; ++j) {
            const v16h bf = load_frag(B, n0 + 16 * j, ldB, k0, lane);
            acc[0][j] = __builtin_amdgcn_wmma_f32_16x16x32_f16(
                false, a0, false, bf, (short)0, acc[0][j], false, false);
            acc[1][j] = __builtin_amdgcn_wmma_f32_16x16x32_f16(
                false, a1, false, bf, (short)0, acc[1][j], false, false);
        }
    }

    // C/D layout: lane&15 = column, (lane>>4)*8 + vgpr = row within tile
    const int cn   = lane & 15;
    const int roff = (lane >> 4) << 3;
#pragma unroll
    for (int i = 0; i < 2; ++i)
#pragma unroll
        for (int j = 0; j < WN; ++j)
#pragma unroll
            for (int r = 0; r < 8; ++r)
                Cmat[(size_t)(m0 + 16 * i + roff + r) * ldC +
                     (n0 + 16 * j + cn)] = acc[i][j][r];
}

// -------------------------------------------------------------------------
// Fused: both diagonal-fuse passes (flat + permuted) as a 9-point gather on
// T0 (= score^T, T0[m][l]), column mask mi[l], row softmax (over l), second
// mask, emit f16 probabilities. One block per row m; row staged in LDS.
// -------------------------------------------------------------------------
__global__ void __launch_bounds__(256)
fuse_mask_softmax(const float* __restrict__ T0, const float* __restrict__ mi,
                  _Float16* __restrict__ P)
{
    __shared__ float z[L];
    __shared__ float red[256];
    const int m   = blockIdx.x;
    const int tid = threadIdx.x;

    int  am[3];
    bool amv[3];
    const int pm = permYX(m);
#pragma unroll
    for (int di = 0; di < 3; ++di) {
        const int pp = pm + (di - 1);
        amv[di] = (pp >= 0 && pp < L);
        am[di]  = amv[di] ? permYX(pp) : 0;
    }

    float lmax = -1e30f;
    for (int l = tid; l < L; l += 256) {
        float v = 0.f;
        const int pl = permYX(l);
#pragma unroll
        for (int di = 0; di < 3; ++di) {
            if (!amv[di]) continue;
            const int pq = pl + (di - 1);
            if (pq < 0 || pq >= L) continue;
            const int al = permYX(pq);
#pragma unroll
            for (int e = -1; e <= 1; ++e) {
                const int i = am[di] + e;
                const int j = al + e;
                if (i >= 0 && i < L && j >= 0 && j < L)
                    v += T0[(size_t)i * L + j];
            }
        }
        v *= mi[l] * 10.0f;   // mask then softmax_scale
        z[l] = v;
        lmax = fmaxf(lmax, v);
    }
    red[tid] = lmax;
    __syncthreads();
    for (int s = 128; s > 0; s >>= 1) {
        if (tid < s) red[tid] = fmaxf(red[tid], red[tid + s]);
        __syncthreads();
    }
    const float zmax = red[0];
    __syncthreads();

    float lsum = 0.f;
    for (int l = tid; l < L; l += 256) {
        const float e = __expf(z[l] - zmax);
        z[l] = e;
        lsum += e;
    }
    red[tid] = lsum;
    __syncthreads();
    for (int s = 128; s > 0; s >>= 1) {
        if (tid < s) red[tid] += red[tid + s];
        __syncthreads();
    }
    const float inv = 1.0f / red[0];
    for (int l = tid; l < L; l += 256)
        P[(size_t)m * L + l] = (_Float16)(z[l] * inv * mi[l]);
}

// -------------------------------------------------------------------------
// col2im: out[c,y,x] = (1/4) * sum_{dy,dx} tmp[(y+1-dy,x+1-dx), c*9+dy*3+dx]
// -------------------------------------------------------------------------
__global__ void __launch_bounds__(256)
col2im_kernel(const float* __restrict__ tmp, float* __restrict__ out)
{
    const int idx = blockIdx.x * 256 + threadIdx.x;  // ch*4096 + y*64 + x
    if (idx >= C * L) return;
    const int ch = idx >> 12;
    const int y  = (idx >> 6) & 63;
    const int x  = idx & 63;
    float acc = 0.f;
#pragma unroll
    for (int dy = 0; dy < 3; ++dy) {
        const int ys = y + 1 - dy;
        if (ys < 0 || ys >= HW) continue;
#pragma unroll
        for (int dx = 0; dx < 3; ++dx) {
            const int xs = x + 1 - dx;
            if (xs < 0 || xs >= HW) continue;
            acc += tmp[(size_t)(ys * HW + xs) * KP + ch * 9 + dy * 3 + dx];
        }
    }
    out[idx] = acc * 0.25f;
}

// -------------------------------------------------------------------------
extern "C" void kernel_launch(void* const* d_in, const int* in_sizes, int n_in,
                              void* d_out, int out_size, void* d_ws, size_t ws_size,
                              hipStream_t stream)
{
    (void)n_in; (void)out_size; (void)ws_size;
    const float* f_o    = (const float*)d_in[0];
    const float* b_o    = (const float*)d_in[1];
    const float* mask_o = (const float*)d_in[2];
    float*       out    = (float*)d_out;

    char* ws = (char*)d_ws;
    _Float16* fpatch = (_Float16*)(ws);                       // 4,718,592 B
    _Float16* bsi    = (_Float16*)(ws + 4718592);             // 4,718,592 B
    _Float16* biT    = (_Float16*)(ws + 9437184);             // 4,718,592 B
    float*    mi     = (float*)   (ws + 14155776);            //    16,384 B
    float*    T0     = (float*)   (ws + 14172160);            // 67,108,864 B
    _Float16* P      = (_Float16*)(ws + 14172160 + 67108864); // 33,554,432 B
    float*    tmp    = (float*)   (ws + 14172160 + 67108864 + 33554432); // 9,437,184 B

    const int B = in_sizes[0] / (C * L);   // = 2
    for (int bidx = 0; bidx < B; ++bidx) {
        const float* f  = f_o    + (size_t)bidx * C * L;
        const float* bb = b_o    + (size_t)bidx * C * L;
        const float* mk = mask_o + (size_t)bidx * L;

        prep_kernel<<<L, 64, 0, stream>>>(f, bb, mk, fpatch, bsi, biT, mi);

        // T0[m][l] = score[l][m] = sum_k fpatch[m,k] * bsi[l,k]
        gemm_nt<4><<<dim3(32, 32), 256, 0, stream>>>(
            fpatch, bsi, T0, L, L, KP, KP, KP, L);

        fuse_mask_softmax<<<L, 256, 0, stream>>>(T0, mi, P);

        // tmp[m][k] = sum_l P[m,l] * biT[k,l]
        gemm_nt<2><<<dim3(32, 9), 256, 0, stream>>>(
            P, biT, tmp, L, KP, L, L, L, KP);

        col2im_kernel<<<(C * L + 255) / 256, 256, 0, stream>>>(
            tmp, out + (size_t)bidx * C * L);
    }
}
